// MultiHeadAttention_54365696033338
// MI455X (gfx1250) — matrix operand order repaired and checked by reading, not compiled, not run on hardware
//
#include <hip/hip_runtime.h>

#define B_  4
#define S_  2048
#define DX_ 1024
#define DM_ 1024
#define H_  16
#define DK_ 64

typedef __attribute__((ext_vector_type(16))) __bf16       v16bf;
typedef __attribute__((ext_vector_type(8)))  float        v8f;
typedef __attribute__((ext_vector_type(4)))  unsigned int v4u;
typedef __attribute__((ext_vector_type(2)))  unsigned int v2u;
typedef __attribute__((ext_vector_type(4)))  float        v4f;
typedef __attribute__((ext_vector_type(8)))  int          i32x8;
typedef __attribute__((ext_vector_type(4)))  int          i32x4;

#if defined(__has_builtin)
#if __has_builtin(__builtin_amdgcn_tensor_load_to_lds)
#define HAVE_TDM 1
#endif
#endif
#ifndef HAVE_TDM
#define HAVE_TDM 0
#endif

union Frag {
  v4u   u[2];
  v16bf b;
};

__device__ __forceinline__ unsigned short f2bf(float f) {
  unsigned int x = __float_as_uint(f);
  x += 0x7fffu + ((x >> 16) & 1u);  // round-to-nearest-even
  return (unsigned short)(x >> 16);
}

__device__ __forceinline__ unsigned int f2bf_pk(float lo, float hi) {
  return (unsigned int)f2bf(lo) | ((unsigned int)f2bf(hi) << 16);
}

__device__ __forceinline__ v8f wmma_bf16(const Frag& a, const Frag& b, v8f c) {
  return __builtin_amdgcn_wmma_f32_16x16x32_bf16(
      false, a.b, false, b.b, (short)0, c, false, false);
}

#if HAVE_TDM
// 1-D contiguous global->LDS DMA of `nelem` bf16 elements via the Tensor Data
// Mover.  D# layout per CDNA5 ISA.
__device__ __forceinline__ void tdm_load_1d(const unsigned short* gsrc,
                                            unsigned lds_byte_off,
                                            unsigned nelem) {
  unsigned long long ga = (unsigned long long)(uintptr_t)gsrc;
  v4u g0 = {0u, 0u, 0u, 0u};
  g0[0] = 0x1u;                                        // count=1, user D#
  g0[1] = lds_byte_off;                                // lds_addr (bytes)
  g0[2] = (unsigned)(ga & 0xffffffffu);                // global_addr[31:0]
  g0[3] = (unsigned)((ga >> 32) & 0x1ffffffu) | (2u << 30);  // [56:32] | type=2
  i32x8 g1 = {0, 0, 0, 0, 0, 0, 0, 0};
  g1[0] = (int)(1u << 16);          // workgroup_mask=0, data_size=1 (2 bytes)
  g1[1] = (int)(nelem << 16);       // tensor_dim0[15:0]
  g1[2] = (int)((nelem >> 16) | (1u << 16));  // tensor_dim0 hi | tensor_dim1=1
  g1[3] = (int)(nelem << 16);       // tile_dim0
  g1[4] = 1;                        // tile_dim1=1, tile_dim2=0
  g1[5] = (int)nelem;               // tensor_dim0_stride[31:0]
  i32x4 gz = {0, 0, 0, 0};
#if __clang_major__ >= 23
  i32x8 gz8 = {0, 0, 0, 0, 0, 0, 0, 0};
  __builtin_amdgcn_tensor_load_to_lds(g0, g1, gz, gz, gz8, 0);
#else
  __builtin_amdgcn_tensor_load_to_lds(g0, g1, gz, gz, 0);
#endif
}
#endif

// ---------------------------------------------------------------------------
// Kernel 1: X[8192,1024]f32 @ W[1024,1024]f32 + bias -> dst bf16 [B,H,S,DK]
// Block tile 128(M) x 128(N), 8 waves; double-buffered 64-deep K chunks.
// ---------------------------------------------------------------------------
__global__ __launch_bounds__(256) void proj_kernel(
    const float* __restrict__ X, const float* __restrict__ W,
    const float* __restrict__ bias, unsigned short* __restrict__ dst) {
  __shared__ __align__(16) unsigned short Abf[2][128 * 64];  // A, row-major
  __shared__ __align__(16) unsigned short Bt[2][128 * 64];   // W, [n][k]

  const int tid = threadIdx.x;
  const int w = tid >> 5, lane = tid & 31, hlf = lane >> 4, l16 = lane & 15;
  const int mbase = blockIdx.x * 128;
  const int n0 = blockIdx.y * 128;

  const v8f z8 = {0.f, 0.f, 0.f, 0.f, 0.f, 0.f, 0.f, 0.f};
  v8f acc[8] = {z8, z8, z8, z8, z8, z8, z8, z8};

  // staging helper (KC = 64): A 128x64 f32->bf16, W chunk 64x128 -> Bt[n][k]
  auto stage = [&](int kc, int buf) {
#pragma unroll
    for (int j = 0; j < 8; ++j) {
      int slot = tid + 256 * j;  // 0..2047 float4 slots
      int row = slot >> 4, c4 = slot & 15;
      v4f v = *(const v4f*)(X + (size_t)(mbase + row) * DX_ + kc + c4 * 4);
      v2u pk = {f2bf_pk(v.x, v.y), f2bf_pk(v.z, v.w)};
      *(v2u*)(&Abf[buf][row * 64 + c4 * 4]) = pk;
    }
    for (int j = 0; j < 32; ++j) {
      int slot = tid + 256 * j;  // 0..8191
      int kk = slot >> 7, n = slot & 127;
      Bt[buf][n * 64 + kk] = f2bf(W[(size_t)(kc + kk) * DM_ + n0 + n]);
    }
  };

  stage(0, 0);
  for (int i = 0; i < DX_ / 64; ++i) {
    __syncthreads();  // buf i ready; buf i^1 free (prev compute finished)
    if (i + 1 < DX_ / 64) stage((i + 1) * 64, (i + 1) & 1);

    const unsigned short* Ab  = &Abf[i & 1][0];
    const unsigned short* Btb = &Bt[i & 1][0];
#pragma unroll
    for (int ks = 0; ks < 2; ++ks) {
      Frag a;
      const v4u* arow = (const v4u*)(Ab + (w * 16 + l16) * 64 + ks * 32);
      a.u[0] = arow[hlf];
      a.u[1] = arow[hlf + 2];
#pragma unroll
      for (int nt = 0; nt < 8; ++nt) {
        Frag bfr;
        const v4u* brow = (const v4u*)(Btb + (nt * 16 + l16) * 64 + ks * 32);
        bfr.u[0] = brow[hlf];
        bfr.u[1] = brow[hlf + 2];
        acc[nt] = wmma_bf16(a, bfr, acc[nt]);
      }
    }
  }

#pragma unroll
  for (int nt = 0; nt < 8; ++nt) {
    int n = n0 + nt * 16 + l16;
    int h = n >> 6, dk = n & 63;
    float bv = bias[n];
#pragma unroll
    for (int r = 0; r < 8; ++r) {
      int m = mbase + w * 16 + r + 8 * hlf;
      int bb = m >> 11, s = m & (S_ - 1);
      dst[(((size_t)(bb * H_ + h)) * S_ + s) * DK_ + dk] = f2bf(acc[nt][r] + bv);
    }
  }
}

// ---------------------------------------------------------------------------
// Kernel 2: flash attention per (b,h,128 q-rows). 8 waves x 16 q-rows.
// K tiles double-buffered via pipelined TDM DMA; V staged transposed;
// P re-materialized through per-wave LDS.
// ---------------------------------------------------------------------------
__global__ __launch_bounds__(256) void attn_kernel(
    const unsigned short* __restrict__ Qh, const unsigned short* __restrict__ Kh,
    const unsigned short* __restrict__ Vh, unsigned short* __restrict__ Z) {
  __shared__ __align__(16) unsigned short Klds[2][32 * 64];   // [key][dk] x2
  __shared__ __align__(16) unsigned short Vt[64 * 32];        // [dk][key]
  __shared__ __align__(16) unsigned short Plds[8 * 16 * 32];  // per-wave P slab

  const int tid = threadIdx.x;
  const int w = tid >> 5, lane = tid & 31, hlf = lane >> 4, l16 = lane & 15;
  const int qb = blockIdx.x * 128;
  const int h = blockIdx.y, b = blockIdx.z;

  const size_t headoff = ((size_t)(b * H_ + h)) * S_ * DK_;
  const unsigned short* Qp = Qh + headoff;
  const unsigned short* Kp = Kh + headoff;
  const unsigned short* Vp = Vh + headoff;
  unsigned short* Pw = &Plds[w * 16 * 32];

  const int qrow = qb + w * 16 + l16;
  Frag qf[2];
#pragma unroll
  for (int kc2 = 0; kc2 < 2; ++kc2) {
    const unsigned short* base = Qp + (size_t)qrow * DK_ + kc2 * 32 + hlf * 8;
    qf[kc2].u[0] = *(const v4u*)base;
    qf[kc2].u[1] = *(const v4u*)(base + 16);
  }

  const v8f z8 = {0.f, 0.f, 0.f, 0.f, 0.f, 0.f, 0.f, 0.f};
  v8f ov[4] = {z8, z8, z8, z8};
  float mr[8], lr[8];
#pragma unroll
  for (int r = 0; r < 8; ++r) { mr[r] = -1e30f; lr[r] = 0.f; }

  const int nkt = qb / 32 + 4;  // causal extent of this 128-row block
#if HAVE_TDM
  if (w == 0)  // prime the pipeline: async DMA of K tile 0
    tdm_load_1d(Kp, (unsigned)(uintptr_t)(void*)(&Klds[0][0]), 32 * DK_);
#endif

  for (int kt = 0; kt < nkt; ++kt) {
    {  // stage V tile transposed: Vt[dk][key]; prefetch next V tile
      const unsigned short* vsrc = Vp + (size_t)kt * 32 * DK_;
      if (kt + 1 < nkt)
        __builtin_prefetch(vsrc + 32 * DK_ + tid * 8, 0, 0);
#pragma unroll
      for (int j = 0; j < 8; ++j) {
        int idx = tid + 256 * j;
        int dk = idx & 63, key = idx >> 6;
        Vt[dk * 32 + key] = vsrc[idx];
      }
    }
#if HAVE_TDM
    if (w == 0) {  // pipelined TDM: issue kt+1, then wait for kt only
      if (kt + 1 < nkt) {
        tdm_load_1d(Kp + (size_t)(kt + 1) * 32 * DK_,
                    (unsigned)(uintptr_t)(void*)(&Klds[(kt + 1) & 1][0]),
                    32 * DK_);
        __builtin_amdgcn_s_wait_tensorcnt(1);  // in-order: tile kt complete
      } else {
        __builtin_amdgcn_s_wait_tensorcnt(0);
      }
    }
#else
    {  // manual K tile copy
      const v4u* src = (const v4u*)(Kp + (size_t)kt * 32 * DK_);
      ((v4u*)&Klds[kt & 1][0])[tid] = src[tid];
    }
#endif
    __syncthreads();

    // scores: preload 4 K-frags, then 4 WMMAs with interleaved accumulators
    const unsigned short* Kb = &Klds[kt & 1][0];
    Frag kf[2][2];
#pragma unroll
    for (int nt = 0; nt < 2; ++nt)
#pragma unroll
      for (int kc2 = 0; kc2 < 2; ++kc2) {
        const unsigned short* kb_ =
            Kb + (nt * 16 + l16) * DK_ + kc2 * 32 + hlf * 8;
        kf[nt][kc2].u[0] = *(const v4u*)kb_;
        kf[nt][kc2].u[1] = *(const v4u*)(kb_ + 16);
      }
    v8f sacc[2] = {z8, z8};
    sacc[0] = wmma_bf16(qf[0], kf[0][0], sacc[0]);
    sacc[1] = wmma_bf16(qf[0], kf[1][0], sacc[1]);
    sacc[0] = wmma_bf16(qf[1], kf[0][1], sacc[0]);
    sacc[1] = wmma_bf16(qf[1], kf[1][1], sacc[1]);

    // online softmax over the 32-key tile
    const int kg0 = kt * 32 + l16;
#pragma unroll
    for (int r = 0; r < 8; ++r) {
      const int q = qb + w * 16 + r + 8 * hlf;
      float s0 = sacc[0][r] * 0.125f;  // 1/sqrt(64)
      float s1 = sacc[1][r] * 0.125f;
      if (kg0 > q)      s0 = -1e30f;   // causal mask
      if (kg0 + 16 > q) s1 = -1e30f;
      float mx = fmaxf(s0, s1);
#pragma unroll
      for (int d = 1; d < 16; d <<= 1) mx = fmaxf(mx, __shfl_xor(mx, d, 32));
      float mnew = fmaxf(mr[r], mx);
      float al = __expf(mr[r] - mnew);
      float p0 = __expf(s0 - mnew);
      float p1 = __expf(s1 - mnew);
      float rs = p0 + p1;
#pragma unroll
      for (int d = 1; d < 16; d <<= 1) rs += __shfl_xor(rs, d, 32);
      lr[r] = lr[r] * al + rs;
      mr[r] = mnew;
#pragma unroll
      for (int t = 0; t < 4; ++t) ov[t][r] *= al;
      Pw[(r + 8 * hlf) * 32 + l16]      = f2bf(p0);
      Pw[(r + 8 * hlf) * 32 + 16 + l16] = f2bf(p1);
    }

    asm volatile("s_wait_dscnt 0" ::: "memory");  // wave-local P RAW fence

    Frag pf;  // A-frag of P
    const unsigned short* pb = Pw + l16 * 32 + hlf * 8;
    pf.u[0] = *(const v4u*)pb;
    pf.u[1] = *(const v4u*)(pb + 16);
#pragma unroll
    for (int t = 0; t < 4; ++t) {
      Frag vf;
      const unsigned short* vb = &Vt[(t * 16 + l16) * 32 + hlf * 8];
      vf.u[0] = *(const v4u*)vb;
      vf.u[1] = *(const v4u*)(vb + 16);
      ov[t] = wmma_bf16(pf, vf, ov[t]);
    }
    __syncthreads();  // protect Vt (+ K buffer reuse two tiles later)
  }

  float inv[8];
#pragma unroll
  for (int r = 0; r < 8; ++r) inv[r] = 1.0f / lr[r];
#pragma unroll
  for (int t = 0; t < 4; ++t)
#pragma unroll
    for (int r = 0; r < 8; ++r) {
      int q = qb + w * 16 + r + 8 * hlf;
      Z[((size_t)(b * S_ + q)) * DM_ + h * DK_ + t * 16 + l16] =
          f2bf(ov[t][r] * inv[r]);
    }
}

// ---------------------------------------------------------------------------
// Kernel 3: Z[8192,1024]bf16 @ wo[1024,1024]f32 + bo -> out f32, tile 128x128
// Double-buffered 64-deep K chunks; A-frags straight from global bf16.
// ---------------------------------------------------------------------------
__global__ __launch_bounds__(256) void oproj_kernel(
    const unsigned short* __restrict__ Z, const float* __restrict__ W,
    const float* __restrict__ bias, float* __restrict__ out) {
  __shared__ __align__(16) unsigned short Bt[2][128 * 64];

  const int tid = threadIdx.x;
  const int w = tid >> 5, lane = tid & 31, hlf = lane >> 4, l16 = lane & 15;
  const int mbase = blockIdx.x * 128;
  const int n0 = blockIdx.y * 128;
  const int mrow = mbase + w * 16 + l16;

  const v8f z8 = {0.f, 0.f, 0.f, 0.f, 0.f, 0.f, 0.f, 0.f};
  v8f acc[8] = {z8, z8, z8, z8, z8, z8, z8, z8};

  auto stage = [&](int kc, int buf) {
    for (int j = 0; j < 32; ++j) {
      int slot = tid + 256 * j;  // 0..8191
      int kk = slot >> 7, n = slot & 127;
      Bt[buf][n * 64 + kk] = f2bf(W[(size_t)(kc + kk) * DX_ + n0 + n]);
    }
  };

  stage(0, 0);
  for (int i = 0; i < DM_ / 64; ++i) {
    __syncthreads();
    if (i + 1 < DM_ / 64) stage((i + 1) * 64, (i + 1) & 1);

    const unsigned short* Btb = &Bt[i & 1][0];
    const int kc = i * 64;
#pragma unroll
    for (int ks = 0; ks < 2; ++ks) {
      Frag a;
      const unsigned short* ab =
          Z + (size_t)mrow * DM_ + kc + ks * 32 + hlf * 8;
      a.u[0] = *(const v4u*)ab;
      a.u[1] = *(const v4u*)(ab + 16);
#pragma unroll
      for (int nt = 0; nt < 8; ++nt) {
        Frag bfr;
        const v4u* brow = (const v4u*)(Btb + (nt * 16 + l16) * 64 + ks * 32);
        bfr.u[0] = brow[hlf];
        bfr.u[1] = brow[hlf + 2];
        acc[nt] = wmma_bf16(a, bfr, acc[nt]);
      }
    }
  }

#pragma unroll
  for (int nt = 0; nt < 8; ++nt) {
    int n = n0 + nt * 16 + l16;
    float bv = bias[n];
#pragma unroll
    for (int r = 0; r < 8; ++r) {
      int m = mbase + w * 16 + r + 8 * hlf;
      out[(size_t)m * DX_ + n] = acc[nt][r] + bv;
    }
  }
}

// ---------------------------------------------------------------------------
extern "C" void kernel_launch(void* const* d_in, const int* in_sizes, int n_in,
                              void* d_out, int out_size, void* d_ws, size_t ws_size,
                              hipStream_t stream) {
  const float* q  = (const float*)d_in[0];
  const float* k  = (const float*)d_in[1];
  const float* v  = (const float*)d_in[2];
  const float* wq = (const float*)d_in[3];
  const float* bq = (const float*)d_in[4];
  const float* wk = (const float*)d_in[5];
  const float* bk = (const float*)d_in[6];
  const float* wv = (const float*)d_in[7];
  const float* bv = (const float*)d_in[8];
  const float* wo = (const float*)d_in[9];
  const float* bo = (const float*)d_in[10];
  // d_in[11] is the causal mask; applied analytically in attn_kernel.
  float* out = (float*)d_out;

  const size_t nhd = (size_t)B_ * H_ * S_ * DK_;
  unsigned short* Qh = (unsigned short*)d_ws;
  unsigned short* Kh = Qh + nhd;
  unsigned short* Vh = Kh + nhd;
  unsigned short* Zs = Vh + nhd;

  dim3 gp(64, 8);  // (8192/128, 1024/128)
  proj_kernel<<<gp, 256, 0, stream>>>(q, wq, bq, Qh);
  proj_kernel<<<gp, 256, 0, stream>>>(k, wk, bk, Kh);
  proj_kernel<<<gp, 256, 0, stream>>>(v, wv, bv, Vh);

  dim3 ga(S_ / 128, H_, B_);
  attn_kernel<<<ga, 256, 0, stream>>>(Qh, Kh, Vh, Zs);

  dim3 go(64, 8);
  oproj_kernel<<<go, 256, 0, stream>>>(Zs, wo, bo, out);
}
